// PatchEmbedding_45707041964447
// MI455X (gfx1250) — 8-bit matrix operand order repaired and checked by reading, compiled, not run on hardware
//
#include <hip/hip_runtime.h>

typedef __attribute__((ext_vector_type(8))) int   v8i;
typedef __attribute__((ext_vector_type(4))) int   v4i;
typedef __attribute__((ext_vector_type(2))) int   v2i;
typedef __attribute__((ext_vector_type(4))) float v4f;

#define B_      16
#define P_      9216
#define D_      768
#define H_      768
#define M_TOTAL (B_ * P_)
#define K_TOTAL D_
#define N_TOTAL H_

#define TILE_M 128
#define TILE_N 128
#define TILE_K 64
#define KSTEPS (K_TOTAL / TILE_K)
#define LDS_STRIDE 80

__device__ __forceinline__ unsigned lds_off32(const void* p) {
  return (unsigned)(size_t)p;
}

__device__ __forceinline__ void async_copy_b128(unsigned lds_addr,
                                                const void* gaddr) {
  asm volatile("global_load_async_to_lds_b128 %0, %1, off"
               :
               : "v"(lds_addr), "v"(gaddr)
               : "memory");
}

__device__ __forceinline__ void wait_asynccnt0() {
  asm volatile("s_wait_asynccnt 0" ::: "memory");
}

__global__ void maxabs_kernel(const float* __restrict__ w,
                              unsigned* __restrict__ maxbits, int n) {
  float m = 0.0f;
  for (int i = blockIdx.x * blockDim.x + threadIdx.x; i < n;
       i += gridDim.x * blockDim.x)
    m = fmaxf(m, fabsf(w[i]));
  #pragma unroll
  for (int off = 16; off > 0; off >>= 1)
    m = fmaxf(m, __shfl_xor(m, off, 32));
  if ((threadIdx.x & 31) == 0)
    atomicMax(maxbits, __float_as_uint(m));
}

__global__ void quantw_kernel(const float* __restrict__ w,
                              const unsigned* __restrict__ maxbits,
                              int* __restrict__ wq_packed, int n4) {
  int i = blockIdx.x * blockDim.x + threadIdx.x;
  if (i >= n4) return;
  const float r = 127.0f / __uint_as_float(*maxbits);
  const v4f v = *(const v4f*)(w + (size_t)i * 4);
  int packed = 0;
  #pragma unroll
  for (int j = 0; j < 4; ++j) {
    float qf = fminf(fmaxf(__builtin_rintf(v[j] * r), -128.0f), 127.0f);
    packed |= ((int)qf & 0xFF) << (8 * j);
  }
  wq_packed[i] = packed;
}

__global__ void __launch_bounds__(256)
gemm_kernel(const float* __restrict__ pixel, const char* __restrict__ wq,
            const float* __restrict__ pos, const unsigned* __restrict__ maxbits,
            float* __restrict__ out) {
  __shared__ __align__(16) char ldsA[2][TILE_M * LDS_STRIDE];
  __shared__ __align__(16) char ldsB[2][TILE_N * LDS_STRIDE];

  const int tid    = threadIdx.x;
  const int lane   = tid & 31;
  const int wave   = tid >> 5;
  const int wave_m = wave & 1;
  const int wave_n = wave >> 1;
  const int lrow   = lane & 15;
  const int half   = lane >> 4;

  const int n0 = blockIdx.x * TILE_N;
  const int m0 = blockIdx.y * TILE_M;

  const int arow  = tid >> 1;
  const int khalf = tid & 1;

  v8i acc[4][2];
  {
    v8i z = {};
    #pragma unroll
    for (int t = 0; t < 4; ++t)
      #pragma unroll
      for (int u = 0; u < 2; ++u) acc[t][u] = z;
  }

  v4i aPack[2];

  auto issueAsyncB = [&](int kt, int buf) {
    const char* gb =
        wq + (size_t)(n0 + arow) * K_TOTAL + (size_t)kt * TILE_K + khalf * 32;
    const unsigned lb = lds_off32(&ldsB[buf][arow * LDS_STRIDE + khalf * 32]);
    async_copy_b128(lb, gb);
    async_copy_b128(lb + 16, gb + 16);
  };

  auto loadQuantA = [&](int kt) {
    const float* ga =
        pixel + (size_t)(m0 + arow) * K_TOTAL + (size_t)kt * TILE_K + khalf * 32;
    #pragma unroll
    for (int g = 0; g < 8; ++g) {
      v4f f = *(const v4f*)(ga + g * 4);
      int q = 0;
      #pragma unroll
      for (int j = 0; j < 4; ++j) {
        float qf = fminf(
            fmaxf(__builtin_rintf(__builtin_fmaf(f[j], 254.0f, -127.0f)),
                  -128.0f),
            127.0f);
        q |= ((int)qf & 0xFF) << (8 * j);
      }
      aPack[g >> 2][g & 3] = q;
    }
  };

  auto storeA = [&](int buf) {
    char* la = &ldsA[buf][arow * LDS_STRIDE + khalf * 32];
    *(v4i*)(la)      = aPack[0];
    *(v4i*)(la + 16) = aPack[1];
  };

  auto compute = [&](int buf) {
    v8i a[4], b[2];
    #pragma unroll
    for (int t = 0; t < 4; ++t) {
      const char* p =
          &ldsA[buf][(wave_m * 64 + t * 16 + lrow) * LDS_STRIDE + half * 8];
      v2i x0 = *(const v2i*)(p);
      v2i x1 = *(const v2i*)(p + 16);
      v2i x2 = *(const v2i*)(p + 32);
      v2i x3 = *(const v2i*)(p + 48);
      v8i av;
      av[0] = x0.x; av[1] = x0.y; av[2] = x1.x; av[3] = x1.y;
      av[4] = x2.x; av[5] = x2.y; av[6] = x3.x; av[7] = x3.y;
      a[t] = av;
    }
    #pragma unroll
    for (int u = 0; u < 2; ++u) {
      const char* p =
          &ldsB[buf][(wave_n * 32 + u * 16 + lrow) * LDS_STRIDE + half * 8];
      v2i y0 = *(const v2i*)(p);
      v2i y1 = *(const v2i*)(p + 16);
      v2i y2 = *(const v2i*)(p + 32);
      v2i y3 = *(const v2i*)(p + 48);
      v8i bv;
      bv[0] = y0.x; bv[1] = y0.y; bv[2] = y1.x; bv[3] = y1.y;
      bv[4] = y2.x; bv[5] = y2.y; bv[6] = y3.x; bv[7] = y3.y;
      b[u] = bv;
    }
    #pragma unroll
    for (int t = 0; t < 4; ++t)
      #pragma unroll
      for (int u = 0; u < 2; ++u)
        acc[t][u] = __builtin_amdgcn_wmma_i32_16x16x64_iu8(
            true, a[t], true, b[u], acc[t][u], false, false);
  };

  issueAsyncB(0, 0);
  loadQuantA(0);
  storeA(0);
  wait_asynccnt0();
  __syncthreads();

  #pragma unroll 1
  for (int kt = 0; kt < KSTEPS; ++kt) {
    const int cur = kt & 1;
    const int nxt = cur ^ 1;
    if (kt + 1 < KSTEPS) {
      issueAsyncB(kt + 1, nxt);
      __builtin_prefetch(
          pixel + (size_t)(m0 + arow) * K_TOTAL + (size_t)(kt + 1) * TILE_K,
          0, 3);
      loadQuantA(kt + 1);
      storeA(nxt);
    }
    compute(cur);
    wait_asynccnt0();
    __syncthreads();
  }

  const float s = __uint_as_float(*maxbits) * (1.0f / 16129.0f);

  #pragma unroll
  for (int t = 0; t < 4; ++t) {
    #pragma unroll
    for (int u = 0; u < 2; ++u) {
      const int gcol = n0 + wave_n * 32 + u * 16 + lrow;
      #pragma unroll
      for (int r = 0; r < 8; ++r) {
        const int grow = m0 + wave_m * 64 + t * 16 + r + half * 8;
        const int prow = grow % P_;
        out[(size_t)grow * N_TOTAL + gcol] =
            (float)acc[t][u][r] * s + pos[(size_t)prow * N_TOTAL + gcol];
      }
    }
  }
}

extern "C" void kernel_launch(void* const* d_in, const int* in_sizes, int n_in,
                              void* d_out, int out_size, void* d_ws,
                              size_t ws_size, hipStream_t stream) {
  const float* pixel = (const float*)d_in[0];
  const float* w     = (const float*)d_in[1];
  const float* pos   = (const float*)d_in[2];
  float* out = (float*)d_out;

  unsigned* maxbits = (unsigned*)d_ws;
  char* wq = (char*)d_ws + 16;

  hipMemsetAsync(d_ws, 0, 16, stream);

  const int nW = H_ * D_;
  maxabs_kernel<<<288, 256, 0, stream>>>(w, maxbits, nW);
  quantw_kernel<<<(nW / 4 + 255) / 256, 256, 0, stream>>>(
      w, maxbits, (int*)wq, nW / 4);

  dim3 grid(N_TOTAL / TILE_N, M_TOTAL / TILE_M);
  gemm_kernel<<<grid, 256, 0, stream>>>(pixel, wq, pos, maxbits, out);
}
